// PCVRSymbiosis_7748121002261
// MI455X (gfx1250) — matrix operand order repaired and checked by reading, compiled, not run on hardware
//
#include <hip/hip_runtime.h>
#include <hip/hip_bf16.h>

// ---------------- problem constants (match reference) ----------------
#define BB   32          // batch
#define NN   4096        // seq len
#define DD   512         // model dim
#define BSS  32          // block size
#define NBB  128         // num blocks (N/BS)
#define HH   8           // index heads
#define IDXD 64          // index dim

typedef __attribute__((ext_vector_type(16))) __bf16 v16bf;
typedef __attribute__((ext_vector_type(8)))  __bf16 bf16x8;
typedef __attribute__((ext_vector_type(8)))  float  f32x8;

// payload type for the async-to-LDS builtin (b128 = 4 dwords)
typedef int v4i_vs __attribute__((vector_size(16)));
typedef __attribute__((address_space(1))) v4i_vs* g_v4i_ptr;
typedef __attribute__((address_space(3))) v4i_vs* l_v4i_ptr;

#define F32_MIN_V (-3.402823466e38f)
#define F32_EPS_V (1.1920929e-7f)
#define EPS_RMS_V (1e-6f)

#if __has_builtin(__builtin_amdgcn_global_load_async_to_lds_b128)
#define HAVE_ASYNC_LDS 1
#else
#define HAVE_ASYNC_LDS 0
#endif

// =====================================================================
// Kernel 0: transpose Wv/Ww (D x D, row = input dim) into bf16 W^T
// (row = output dim n, contiguous K) so WMMA B-fragments are one
// contiguous 32B load per lane.
// =====================================================================
__global__ __launch_bounds__(256) void prep_weights_kernel(
    const float* __restrict__ Wv, const float* __restrict__ Ww,
    __bf16* __restrict__ WvT, __bf16* __restrict__ WwT)
{
    int idx = blockIdx.x * 256 + threadIdx.x;       // 0 .. 2*512*512-1
    const float* W = (idx < DD * DD) ? Wv : Ww;
    __bf16*      O = (idx < DD * DD) ? WvT : WwT;
    int i = idx & (DD * DD - 1);
    int n = i >> 9;            // output dim
    int k = i & (DD - 1);      // input dim
    O[i] = (__bf16)W[k * DD + n];
}

// =====================================================================
// Kernel 1: fused LearnedBlockCompressor.
// grid = B*NB workgroups, 256 threads (8 waves).
// Per WG: X (32x512 tokens) -> LDS (async-to-LDS if available) -> bf16;
// logits GEMM (WMMA bf16, f32 acc); masked column softmax; values GEMM
// fused with weighted sum; rmsnorm.
// =====================================================================
#define XSTR 520   // bf16 row stride for sX; 520*2 = 1040 B = 65*16 (16B aligned rows)

__global__ __launch_bounds__(256) void compress_kernel(
    const float* __restrict__ tokens,
    const unsigned char* __restrict__ pmask,
    const __bf16* __restrict__ WvT,
    const __bf16* __restrict__ WwT,
    const float* __restrict__ bv,
    const float* __restrict__ bw,
    const float* __restrict__ pos_bias,
    const float* __restrict__ g_out,
    float* __restrict__ block_tokens,
    int* __restrict__ block_pad)
{
    __shared__ __attribute__((aligned(64))) __bf16 sX[BSS][XSTR];   // ~33 KB
    // sW doubles as the fp32 async staging buffer (exactly 32*512*4 = 64 KB).
    // Staging is fully consumed (converted into sX) before the logits epilogue
    // writes sW; the phases are separated by __syncthreads().
    __shared__ __attribute__((aligned(64))) float sW[BSS][DD];      // 64 KB
    __shared__ float sPart[2][DD];                                  // per-M-tile weighted sums
    __shared__ float sWs[8];
    __shared__ unsigned char sPad[BSS];
    __shared__ float sSum;
    __shared__ int   sAllPad;

    const int t    = threadIdx.x;
    const int wave = t >> 5;
    const int lane = t & 31;
    const int b    = blockIdx.x >> 7;          // / NB
    const int nb   = blockIdx.x & (NBB - 1);   // % NB

    // L2 warm hint for the (shared) weight streams — global_prefetch_b8
    if (t < 8) {
        __builtin_prefetch(WwT + (size_t)t * 64 * DD, 0, 1);
        __builtin_prefetch(WvT + (size_t)t * 64 * DD, 0, 1);
    }

    // ---------------- phase 1: load X tile -> LDS (bf16), pads ----------------
    if (t < BSS) sPad[t] = pmask[(size_t)b * NN + nb * BSS + t];
    const float* gsrc = tokens + ((size_t)b * NN + (size_t)nb * BSS) * DD;

#if HAVE_ASYNC_LDS
    {
        float* sStage = &sW[0][0];
        // 16384 floats = 4096 x b128; 16 async copies per thread, no VGPR data.
#pragma unroll
        for (int i = 0; i < 16; ++i) {
            int idx4 = t + i * 256;
            __builtin_amdgcn_global_load_async_to_lds_b128(
                (g_v4i_ptr)(uintptr_t)(const void*)(gsrc + idx4 * 4),
                (l_v4i_ptr)(uintptr_t)(void*)(sStage + idx4 * 4),
                0, 0);
        }
#if __has_builtin(__builtin_amdgcn_s_wait_asynccnt)
        __builtin_amdgcn_s_wait_asynccnt(0);
#else
        asm volatile("s_wait_asynccnt 0" ::: "memory");
#endif
    }
    __syncthreads();   // all waves' async LDS writes visible
    // convert staging fp32 -> bf16 A-operand tile
#pragma unroll
    for (int i = 0; i < 16; ++i) {
        int idx4 = t + i * 256;
        int e = idx4 * 4;
        int r = e >> 9;
        int c = e & (DD - 1);
        float4 v = *(const float4*)(&sW[0][0] + e);
        __bf16* dst = &sX[r][c];
        dst[0] = (__bf16)v.x; dst[1] = (__bf16)v.y;
        dst[2] = (__bf16)v.z; dst[3] = (__bf16)v.w;
    }
#else
    {
        const float4* src = reinterpret_cast<const float4*>(gsrc);
#pragma unroll
        for (int i = 0; i < 16; ++i) {
            int idx4 = t + i * 256;            // 0..4095 float4s
            float4 v = src[idx4];
            int e = idx4 * 4;
            int r = e >> 9;
            int c = e & (DD - 1);
            __bf16* dst = &sX[r][c];
            dst[0] = (__bf16)v.x; dst[1] = (__bf16)v.y;
            dst[2] = (__bf16)v.z; dst[3] = (__bf16)v.w;
        }
    }
#endif
    __syncthreads();   // sX ready; staging (sW) free for reuse as logits

    if (t == 0) {
        int ap = 1;
        for (int s = 0; s < BSS; ++s) ap &= (int)(sPad[s] != 0);
        sAllPad = ap;
    }

    // GEMM work assignment: wave -> M tile (0/1) and 8 N tiles
    const int mtile   = wave & 1;
    const int n0      = (wave >> 1) * 8;       // first N tile of this wave
    const int halfhi  = (lane < 16) ? 0 : 8;   // accumulator row offset per lane half (D row 8h + r)
    const int rowA    = mtile * 16 + (lane & 15);
    const int colLane = lane & 15;

    // ---------------- phase 2: logits GEMM (X @ Ww) ----------------
    {
        f32x8 acc[8];
#pragma unroll
        for (int n = 0; n < 8; ++n)
#pragma unroll
            for (int e = 0; e < 8; ++e) acc[n][e] = 0.0f;

        for (int kt = 0; kt < 16; ++kt) {
            int kbA = kt * 32 + ((lane < 16) ? 0 : 16);
            bf16x8 a0 = *(const bf16x8*)&sX[rowA][kbA];
            bf16x8 a1 = *(const bf16x8*)&sX[rowA][kbA + 8];
            v16bf A;
#pragma unroll
            for (int e = 0; e < 8; ++e) { A[e] = a0[e]; A[e + 8] = a1[e]; }
            int kbB = kt * 32 + ((lane < 16) ? 0 : 16);
#pragma unroll
            for (int n = 0; n < 8; ++n) {
                const __bf16* wp = WwT + (size_t)((n0 + n) * 16 + colLane) * DD + kbB;
                v16bf Bf = *(const v16bf*)wp;
                acc[n] = __builtin_amdgcn_wmma_f32_16x16x32_bf16(
                    false, A, false, Bf, (short)0, acc[n], false, false);
            }
        }
        // epilogue: + bw + pos_bias, store raw logits to sW
#pragma unroll
        for (int n = 0; n < 8; ++n) {
            int col = (n0 + n) * 16 + colLane;
            float bwc = bw[col];
#pragma unroll
            for (int r = 0; r < 8; ++r) {
                int gm = mtile * 16 + r + halfhi;     // row within 32-token block
                sW[gm][col] = acc[n][r] + bwc + pos_bias[gm * DD + col];
            }
        }
    }
    __syncthreads();

    // ---------------- phase 3: masked column softmax (over 32 rows) ----------------
#pragma unroll
    for (int cc = 0; cc < 2; ++cc) {
        int col = t + cc * 256;
        float mx = F32_MIN_V;
        for (int s = 0; s < BSS; ++s)
            if (!sPad[s]) mx = fmaxf(mx, sW[s][col]);
        float den = 0.0f;
        for (int s = 0; s < BSS; ++s) {
            float e = sPad[s] ? 0.0f : __expf(sW[s][col] - mx);
            sW[s][col] = e;
            den += e;
        }
        float inv = 1.0f / fmaxf(den, F32_EPS_V);
        for (int s = 0; s < BSS; ++s) sW[s][col] *= inv;
    }
    __syncthreads();

    // ---------------- phase 4: values GEMM (X @ Wv) fused with weighted sum ----------------
    {
        f32x8 acc[8];
#pragma unroll
        for (int n = 0; n < 8; ++n)
#pragma unroll
            for (int e = 0; e < 8; ++e) acc[n][e] = 0.0f;

        for (int kt = 0; kt < 16; ++kt) {
            int kbA = kt * 32 + ((lane < 16) ? 0 : 16);
            bf16x8 a0 = *(const bf16x8*)&sX[rowA][kbA];
            bf16x8 a1 = *(const bf16x8*)&sX[rowA][kbA + 8];
            v16bf A;
#pragma unroll
            for (int e = 0; e < 8; ++e) { A[e] = a0[e]; A[e + 8] = a1[e]; }
            int kbB = kt * 32 + ((lane < 16) ? 0 : 16);
#pragma unroll
            for (int n = 0; n < 8; ++n) {
                const __bf16* wp = WvT + (size_t)((n0 + n) * 16 + colLane) * DD + kbB;
                v16bf Bf = *(const v16bf*)wp;
                acc[n] = __builtin_amdgcn_wmma_f32_16x16x32_bf16(
                    false, A, false, Bf, (short)0, acc[n], false, false);
            }
        }
        // epilogue: out[col] partial = sum_rows w * (v + bv); deterministic combine
#pragma unroll
        for (int n = 0; n < 8; ++n) {
            int col = (n0 + n) * 16 + colLane;
            float bvc = bv[col];
            float ps = 0.0f;
#pragma unroll
            for (int r = 0; r < 8; ++r) {
                int gm = mtile * 16 + r + halfhi;
                ps += sW[gm][col] * (acc[n][r] + bvc);
            }
            // combine lane l (rows 0..7) with lane l+16 (rows 8..15) of this M tile
            ps += __shfl_xor(ps, 16, 32);
            if (lane < 16) sPart[mtile][col] = ps;   // unique (mtile,col) per writer
        }
    }
    __syncthreads();

    // ---------------- phase 5: rmsnorm + write block_tokens ----------------
    float v0 = sPart[0][t]       + sPart[1][t];
    float v1 = sPart[0][t + 256] + sPart[1][t + 256];
    float ss = v0 * v0 + v1 * v1;
#pragma unroll
    for (int o = 16; o > 0; o >>= 1) ss += __shfl_xor(ss, o, 32);
    if (lane == 0) sWs[wave] = ss;
    __syncthreads();
    if (t == 0) {
        float tot = 0.0f;
        for (int w = 0; w < 8; ++w) tot += sWs[w];
        sSum = tot;
    }
    __syncthreads();

    float scale = rsqrtf(sSum * (1.0f / (float)DD) + EPS_RMS_V);
    if (sAllPad) scale = 0.0f;
    float* outp = block_tokens + ((size_t)b * NBB + nb) * DD;
    outp[t]       = v0 * scale * g_out[t];
    outp[t + 256] = v1 * scale * g_out[t + 256];
    if (t == 0) block_pad[b * NBB + nb] = sAllPad;
}

// =====================================================================
// Kernel 2: SparseBlockIndexer + top-k + gather. grid = B, 256 threads.
// =====================================================================
__global__ __launch_bounds__(256) void indexer_kernel(
    const float* __restrict__ query,
    const float* __restrict__ Wqd, const float* __restrict__ bqd,
    const float* __restrict__ Wqu, const float* __restrict__ bqu,
    const float* __restrict__ Wk,  const float* __restrict__ bk,
    const float* __restrict__ Whw, const float* __restrict__ bhw,
    const float* __restrict__ g_q, const float* __restrict__ g_k,
    const float* __restrict__ block_tokens,
    const int* __restrict__ block_pad,
    const int* __restrict__ topk_p,
    float* __restrict__ out)     // [B*keep*D] blocks then [B*keep] mask
{
    __shared__ float sq[DD];
    __shared__ float sql[IDXD];
    __shared__ float sqh[HH * IDXD];
    __shared__ float shw[HH];
    __shared__ float sKeys[NBB][IDXD];
    __shared__ float sScores[NBB];
    __shared__ int   sIdx[NBB];
    __shared__ float srms;

    const int b = blockIdx.x;
    const int t = threadIdx.x;
    const int wave = t >> 5, lane = t & 31;

    sq[t]       = query[(size_t)b * DD + t];
    sq[t + 256] = query[(size_t)b * DD + t + 256];
    __syncthreads();

    // q down-projection + head-weight logits
    if (t < IDXD) {
        float a = bqd[t];
        for (int d = 0; d < DD; ++d) a += sq[d] * Wqd[d * IDXD + t];
        sql[t] = a;
    } else if (t < IDXD + HH) {
        int h = t - IDXD;
        float a = bhw[h];
        for (int d = 0; d < DD; ++d) a += sq[d] * Whw[d * HH + h];
        shw[h] = a;
    }
    __syncthreads();

    if (t == 0) {
        float ssq = 0.0f;
        for (int i = 0; i < IDXD; ++i) ssq += sql[i] * sql[i];
        srms = rsqrtf(ssq * (1.0f / (float)IDXD) + EPS_RMS_V);
        float mx = shw[0];
        for (int h = 1; h < HH; ++h) mx = fmaxf(mx, shw[h]);
        float den = 0.0f, e[HH];
        for (int h = 0; h < HH; ++h) { e[h] = __expf(shw[h] - mx); den += e[h]; }
        for (int h = 0; h < HH; ++h) shw[h] = e[h] / den;
    }
    __syncthreads();
    if (t < IDXD) sql[t] = sql[t] * srms * g_q[t];
    __syncthreads();

    // q heads (up-projection)
    for (int o = t; o < HH * IDXD; o += 256) {
        float a = bqu[o];
        for (int j = 0; j < IDXD; ++j) a += sql[j] * Wqu[j * (HH * IDXD) + o];
        sqh[o] = a;
    }
    __syncthreads();

    // keys: one wave per block-token row, 2 outputs per lane
    const float* bt = block_tokens + (size_t)b * NBB * DD;
    for (int n = wave; n < NBB; n += 8) {
        const float* row = bt + (size_t)n * DD;
        float k0 = bk[lane], k1 = bk[lane + 32];
        for (int d = 0; d < DD; ++d) {
            float x = row[d];
            k0 += x * Wk[d * IDXD + lane];
            k1 += x * Wk[d * IDXD + lane + 32];
        }
        float ssq = k0 * k0 + k1 * k1;
#pragma unroll
        for (int o = 16; o > 0; o >>= 1) ssq += __shfl_xor(ssq, o, 32);
        float sc = rsqrtf(ssq * (1.0f / (float)IDXD) + EPS_RMS_V);
        sKeys[n][lane]      = k0 * sc * g_k[lane];
        sKeys[n][lane + 32] = k1 * sc * g_k[lane + 32];
    }
    __syncthreads();

    // scores = sum_h softmax(head_w)[h] * relu(q_h . key_n), masked
    if (t < NBB) {
        float s = 0.0f;
        for (int h = 0; h < HH; ++h) {
            float a = 0.0f;
            for (int i = 0; i < IDXD; ++i) a += sqh[h * IDXD + i] * sKeys[t][i];
            s += fmaxf(a, 0.0f) * shw[h];
        }
        if (block_pad[b * NBB + t]) s = F32_MIN_V;
        sScores[t] = s;
    }
    __syncthreads();

    int keep = topk_p[0];
    if (keep > NBB) keep = NBB;
    if (keep < 1)   keep = 1;

    // stable top-k (first index wins ties, matching jax.lax.top_k)
    if (t == 0) {
        for (int k = 0; k < keep; ++k) {
            float best = -__builtin_inff();
            int bi = 0;
            for (int n = 0; n < NBB; ++n)
                if (sScores[n] > best) { best = sScores[n]; bi = n; }
            sIdx[k] = bi;
            sScores[bi] = -__builtin_inff();
        }
    }
    __syncthreads();

    // gather selected blocks + mask
    float* mask_out = out + (size_t)BB * keep * DD;
    for (int k = 0; k < keep; ++k) {
        int idx = sIdx[k];
        const float* srcr = bt + (size_t)idx * DD;
        float* dstr = out + ((size_t)b * keep + k) * DD;
        dstr[t]       = srcr[t];
        dstr[t + 256] = srcr[t + 256];
        if (t == 0)
            mask_out[(size_t)b * keep + k] = block_pad[b * NBB + idx] ? 1.0f : 0.0f;
    }
}

// =====================================================================
// Host launcher
// =====================================================================
extern "C" void kernel_launch(void* const* d_in, const int* in_sizes, int n_in,
                              void* d_out, int out_size, void* d_ws, size_t ws_size,
                              hipStream_t stream) {
    const float*         tokens   = (const float*)d_in[0];
    const float*         query    = (const float*)d_in[1];
    const unsigned char* pmask    = (const unsigned char*)d_in[2];
    const int*           topk     = (const int*)d_in[3];
    const float* Wv  = (const float*)d_in[4];
    const float* bv  = (const float*)d_in[5];
    const float* Ww  = (const float*)d_in[6];
    const float* bw  = (const float*)d_in[7];
    const float* pos_bias = (const float*)d_in[8];
    const float* g_out    = (const float*)d_in[9];
    const float* Wqd = (const float*)d_in[10];
    const float* bqd = (const float*)d_in[11];
    const float* Wqu = (const float*)d_in[12];
    const float* bqu = (const float*)d_in[13];
    const float* Wk  = (const float*)d_in[14];
    const float* bk  = (const float*)d_in[15];
    const float* Whw = (const float*)d_in[16];
    const float* bhw = (const float*)d_in[17];
    const float* g_q = (const float*)d_in[18];
    const float* g_k = (const float*)d_in[19];

    char* ws = (char*)d_ws;
    __bf16* WvT          = (__bf16*)ws;                                // 512 KB
    __bf16* WwT          = (__bf16*)(ws + 512 * 1024);                 // 512 KB
    float*  block_tokens = (float*)(ws + 1024 * 1024);                 // 8 MB
    int*    block_pad    = (int*)(ws + 1024 * 1024 + 8 * 1024 * 1024); // 16 KB

    prep_weights_kernel<<<2 * DD * DD / 256, 256, 0, stream>>>(Wv, Ww, WvT, WwT);
    compress_kernel<<<BB * NBB, 256, 0, stream>>>(tokens, pmask, WvT, WwT,
                                                  bv, bw, pos_bias, g_out,
                                                  block_tokens, block_pad);
    indexer_kernel<<<BB, 256, 0, stream>>>(query, Wqd, bqd, Wqu, bqu, Wk, bk,
                                           Whw, bhw, g_q, g_k,
                                           block_tokens, block_pad, topk,
                                           (float*)d_out);
}
